// BertMoEExpertPool_31559419691534
// MI455X (gfx1250) — matrix operand order repaired and checked by reading, not compiled, not run on hardware
//
#include <hip/hip_runtime.h>

// ---- problem constants (from reference) ----
#define NE 8
#define HD 768
#define FD 3072
#define NT 4096

// ---- tiling ----
#define BLK_M 128
#define BLK_N 256
#define BLK_K 32
#define SSTR  40   // LDS row stride in halves (80 B): conflict-free frag reads & DMA stores

typedef __attribute__((ext_vector_type(16))) _Float16 v16h;
typedef __attribute__((ext_vector_type(8)))  _Float16 v8h;
typedef __attribute__((ext_vector_type(4)))  _Float16 v4h;
typedef __attribute__((ext_vector_type(8)))  float    v8f;
typedef __attribute__((ext_vector_type(4)))  float    f32x4;

// tanh-form GELU via CDNA5 hardware transcendental v_tanh_f32.
__device__ __forceinline__ float gelu_tanh(float x) {
    float x2 = x * x;
    float t  = 0.7978845608028654f * x * __builtin_fmaf(0.044715f, x2, 1.0f);
    float th;
    asm("v_tanh_f32 %0, %1" : "=v"(th) : "v"(t));
    return 0.5f * x * (1.0f + th);
}

// async DMA: 16 bytes per lane, global -> LDS, tracked by ASYNCcnt
__device__ __forceinline__ void async_copy_b128(unsigned lds_off, const void* gaddr) {
    asm volatile("global_load_async_to_lds_b128 %0, %1, off"
                 :: "v"(lds_off), "v"((unsigned long long)(size_t)gaddr)
                 : "memory");
}
#define WAIT_ASYNC(N) asm volatile("s_wait_asynccnt " #N ::: "memory")

// ---- pre-pass: straight fp32 -> f16 convert (hidden states) ----
__global__ __launch_bounds__(256)
void cvt_f32_to_f16(const float* __restrict__ src, _Float16* __restrict__ dst, long n)
{
    long i = ((long)blockIdx.x * 256 + threadIdx.x) * 4;
    if (i + 3 < n) {
        f32x4 v = *(const f32x4*)(src + i);
        v4h p;
        p[0] = (_Float16)v[0]; p[1] = (_Float16)v[1];
        p[2] = (_Float16)v[2]; p[3] = (_Float16)v[3];
        *(v4h*)(dst + i) = p;
    }
}

// ---- pre-pass: per-expert transpose + convert: src [E][R][C] f32 -> dst [E][C][R] f16 ----
__global__ __launch_bounds__(256)
void transpose_cvt(const float* __restrict__ src, _Float16* __restrict__ dst, int R, int C)
{
    __shared__ float tile[32][33];
    const int e  = blockIdx.z;
    const int r0 = blockIdx.y * 32;
    const int c0 = blockIdx.x * 32;
    const float* s = src + (size_t)e * (size_t)R * (size_t)C;
    _Float16*    d = dst + (size_t)e * (size_t)R * (size_t)C;
    const int t  = threadIdx.x;
    const int tr = t >> 3;         // 0..31
    const int tc = (t & 7) * 4;    // 0..28

    f32x4 v = *(const f32x4*)(s + (size_t)(r0 + tr) * (size_t)C + (size_t)(c0 + tc));
    tile[tr][tc + 0] = v[0]; tile[tr][tc + 1] = v[1];
    tile[tr][tc + 2] = v[2]; tile[tr][tc + 3] = v[3];
    __syncthreads();

    v4h p;
    p[0] = (_Float16)tile[tc + 0][tr];
    p[1] = (_Float16)tile[tc + 1][tr];
    p[2] = (_Float16)tile[tc + 2][tr];
    p[3] = (_Float16)tile[tc + 3][tr];
    *(v4h*)(d + (size_t)(c0 + tr) * (size_t)R + (size_t)(r0 + tc)) = p;
}

// ---- main GEMM: C[M,Ncols] = A[M,K] * BT[Ncols,K]^T (+bias), optional GELU+f16 out.
// A, BT f16 with row stride == K. Block tile 128x256, 8 waves, each wave a 64x64
// subtile via 4x4 v_wmma_f32_16x16x32_f16. Staging via CDNA5 async global->LDS
// DMA (ASYNCcnt), double-buffered LDS, K-loop unrolled by 2 so buffer indices
// (and all LDS addresses) are compile-time constants. Requires nk even, nk >= 4.
template<bool GELU_F16OUT>
__global__ __launch_bounds__(256, 1)
void moe_gemm_f16(const _Float16* __restrict__ Abase,
                  const _Float16* __restrict__ BTbase,
                  const float* __restrict__ biasBase,
                  void* __restrict__ Cbase,
                  int K, int ldc,
                  long strideA, long strideB, long strideBias, long strideC)
{
    __shared__ __align__(16) _Float16 sA[2][BLK_M * SSTR];   // 2 x (128 rows x 32 k, padded)
    __shared__ __align__(16) _Float16 sB[2][BLK_N * SSTR];   // 2 x (256 cols x 32 k, padded)

    const int tid   = threadIdx.x;
    const int lane  = tid & 31;
    const int wave  = tid >> 5;
    const int wm    = wave >> 2;    // 0..1
    const int wn    = wave & 3;     // 0..3
    const int lrow  = lane & 15;
    const int lhalf = lane >> 4;

    const int e  = blockIdx.z;
    const int m0 = blockIdx.y * BLK_M;
    const int n0 = blockIdx.x * BLK_N;

    const _Float16* A    = Abase    + (size_t)e * (size_t)strideA;
    const _Float16* BT   = BTbase   + (size_t)e * (size_t)strideB;
    const float*    bias = biasBase + (size_t)e * (size_t)strideBias;

    // staging: 4 lanes cover one 64 B row (32 halves); 256 threads = 64 rows/pass
    const int qrow = tid >> 2;         // 0..63
    const int qoff = (tid & 3) * 8;    // half offset 0,8,16,24

    const _Float16* ag0 = A  + (size_t)(m0 + qrow)       * (size_t)K + qoff;
    const _Float16* ag1 = A  + (size_t)(m0 + 64 + qrow)  * (size_t)K + qoff;
    const _Float16* bg0 = BT + (size_t)(n0 + qrow)       * (size_t)K + qoff;
    const _Float16* bg1 = BT + (size_t)(n0 + 64 + qrow)  * (size_t)K + qoff;
    const _Float16* bg2 = BT + (size_t)(n0 + 128 + qrow) * (size_t)K + qoff;
    const _Float16* bg3 = BT + (size_t)(n0 + 192 + qrow) * (size_t)K + qoff;

    // per-lane LDS byte offsets (flat LDS aperture keeps the offset in addr[31:0])
    unsigned la0[2], la1[2], lb0[2], lb1[2], lb2[2], lb3[2];
    #pragma unroll
    for (int b = 0; b < 2; ++b) {
        la0[b] = (unsigned)(size_t)&sA[b][qrow * SSTR + qoff];
        la1[b] = (unsigned)(size_t)&sA[b][(64 + qrow) * SSTR + qoff];
        lb0[b] = (unsigned)(size_t)&sB[b][qrow * SSTR + qoff];
        lb1[b] = (unsigned)(size_t)&sB[b][(64 + qrow) * SSTR + qoff];
        lb2[b] = (unsigned)(size_t)&sB[b][(128 + qrow) * SSTR + qoff];
        lb3[b] = (unsigned)(size_t)&sB[b][(192 + qrow) * SSTR + qoff];
    }

    v8f acc[4][4] = {};
    const int nk = K / BLK_K;          // 24 or 96: even, >= 4

    auto issue_tile = [&](int buf, int kt) {
        const int k0 = kt * BLK_K;
        async_copy_b128(la0[buf], ag0 + k0); async_copy_b128(la1[buf], ag1 + k0);
        async_copy_b128(lb0[buf], bg0 + k0); async_copy_b128(lb1[buf], bg1 + k0);
        async_copy_b128(lb2[buf], bg2 + k0); async_copy_b128(lb3[buf], bg3 + k0);
    };

    auto compute_tile = [&](int buf) {
        const _Float16* cA = sA[buf];
        const _Float16* cB = sB[buf];
        v16h afr[4], bfr[4];
        #pragma unroll
        for (int i = 0; i < 4; ++i) {
            const _Float16* pa = &cA[(wm * 64 + i * 16 + lrow) * SSTR + lhalf * 8];
            v8h lo = *(const v8h*)(pa);
            v8h hi = *(const v8h*)(pa + 16);
            afr[i] = __builtin_shufflevector(lo, hi, 0,1,2,3,4,5,6,7,8,9,10,11,12,13,14,15);
        }
        #pragma unroll
        for (int j = 0; j < 4; ++j) {
            const _Float16* pb = &cB[(wn * 64 + j * 16 + lrow) * SSTR + lhalf * 8];
            v8h lo = *(const v8h*)(pb);
            v8h hi = *(const v8h*)(pb + 16);
            bfr[j] = __builtin_shufflevector(lo, hi, 0,1,2,3,4,5,6,7,8,9,10,11,12,13,14,15);
        }
        #pragma unroll
        for (int i = 0; i < 4; ++i)
            #pragma unroll
            for (int j = 0; j < 4; ++j)
                acc[i][j] = __builtin_amdgcn_wmma_f32_16x16x32_f16(
                    false, afr[i], false, bfr[j], (short)0, acc[i][j], false, false);
    };

    issue_tile(0, 0);

    // main pipeline: branch-free double steps with static buffer indices
    for (int kt = 0; kt < nk - 2; kt += 2) {
        issue_tile(1, kt + 1);
        WAIT_ASYNC(6);               // in-order: tile kt's 6 copies complete
        __syncthreads();
        compute_tile(0);
        __syncthreads();

        issue_tile(0, kt + 2);
        WAIT_ASYNC(6);
        __syncthreads();
        compute_tile(1);
        __syncthreads();
    }

    // tail: tiles nk-2 (buf0) and nk-1 (buf1)
    issue_tile(1, nk - 1);
    WAIT_ASYNC(6);
    __syncthreads();
    compute_tile(0);
    __syncthreads();

    WAIT_ASYNC(0);
    __syncthreads();
    compute_tile(1);

    // ---- epilogue ----
    if (GELU_F16OUT) {
        _Float16* C = (_Float16*)Cbase + (size_t)e * (size_t)strideC;
        #pragma unroll
        for (int j = 0; j < 4; ++j) {
            const int col = n0 + wn * 64 + j * 16 + lrow;
            const float bv = bias[col];
            #pragma unroll
            for (int i = 0; i < 4; ++i) {
                const int rowb = m0 + wm * 64 + i * 16 + lhalf * 8;
                #pragma unroll
                for (int r = 0; r < 8; ++r) {
                    float v = gelu_tanh(acc[i][j][r] + bv);
                    C[(size_t)(rowb + r) * (size_t)ldc + (size_t)col] = (_Float16)v;
                }
            }
        }
    } else {
        float* C = (float*)Cbase + (size_t)e * (size_t)strideC;
        #pragma unroll
        for (int j = 0; j < 4; ++j) {
            const int col = n0 + wn * 64 + j * 16 + lrow;
            const float bv = bias[col];
            #pragma unroll
            for (int i = 0; i < 4; ++i) {
                const int rowb = m0 + wm * 64 + i * 16 + lhalf * 8;
                #pragma unroll
                for (int r = 0; r < 8; ++r)
                    C[(size_t)(rowb + r) * (size_t)ldc + (size_t)col] = acc[i][j][r] + bv;
            }
        }
    }
}

extern "C" void kernel_launch(void* const* d_in, const int* in_sizes, int n_in,
                              void* d_out, int out_size, void* d_ws, size_t ws_size,
                              hipStream_t stream)
{
    const float* hidden = (const float*)d_in[0];   // [N, H]
    const float* Wi     = (const float*)d_in[1];   // [E, H, F]
    const float* bi     = (const float*)d_in[2];   // [E, F]
    const float* Wo     = (const float*)d_in[3];   // [E, F, H]
    const float* bo     = (const float*)d_in[4];   // [E, H]

    // workspace layout (halves)
    _Float16* inter = (_Float16*)d_ws;                           // [E, N, F]
    _Float16* hh    = inter + (size_t)NE * NT * FD;              // [N, H]
    _Float16* WiT   = hh    + (size_t)NT * HD;                   // [E, F, H]  (Wi transposed)
    _Float16* WoT   = WiT   + (size_t)NE * FD * HD;              // [E, H, F]  (Wo transposed)

    // pre-passes: convert (and transpose weights) to f16 once
    cvt_f32_to_f16<<<((long)NT * HD / 4 + 255) / 256, 256, 0, stream>>>(hidden, hh, (long)NT * HD);
    transpose_cvt<<<dim3(FD / 32, HD / 32, NE), 256, 0, stream>>>(Wi, WiT, HD, FD); // [H][F]->[F][H]
    transpose_cvt<<<dim3(HD / 32, FD / 32, NE), 256, 0, stream>>>(Wo, WoT, FD, HD); // [F][H]->[H][F]

    dim3 blk(256, 1, 1);

    // GEMM1: inter[e] = gelu(hidden * Wi[e] + bi[e])   (M=4096, K=768, Ncols=3072)
    dim3 g1(FD / BLK_N, NT / BLK_M, NE);
    moe_gemm_f16<true><<<g1, blk, 0, stream>>>(
        hh, WiT, bi, inter,
        /*K*/HD, /*ldc*/FD,
        /*strideA*/0L, /*strideB*/(long)FD * HD, /*strideBias*/(long)FD,
        /*strideC*/(long)NT * FD);

    // GEMM2: out[e] = inter[e] * Wo[e] + bo[e]         (M=4096, K=3072, Ncols=768)
    dim3 g2(HD / BLK_N, NT / BLK_M, NE);
    moe_gemm_f16<false><<<g2, blk, 0, stream>>>(
        inter, WoT, bo, d_out,
        /*K*/FD, /*ldc*/HD,
        /*strideA*/(long)NT * FD, /*strideB*/(long)HD * FD, /*strideBias*/(long)HD,
        /*strideC*/(long)NT * HD);
}
